// QORNN_Model_64579128262662
// MI455X (gfx1250) — 8-bit matrix operand order repaired and checked by reading, compiled, not run on hardware
//
#include <hip/hip_runtime.h>

typedef __attribute__((ext_vector_type(8))) int v8i;

#define T_STEPS 1024
#define I_DIM   64
#define H_DIM   256
#define O_DIM   16
#define B_TILE  16

__device__ __forceinline__ int qclamp(float v, float scale, int lo, int hi) {
    int i = (int)__builtin_rintf(v * scale);
    i = i < lo ? lo : i;
    i = i > hi ? hi : i;
    return i;
}

__device__ __forceinline__ unsigned int qpack4(float4 f, float scale, int lo, int hi) {
    int a = qclamp(f.x, scale, lo, hi);
    int b = qclamp(f.y, scale, lo, hi);
    int c = qclamp(f.z, scale, lo, hi);
    int d = qclamp(f.w, scale, lo, hi);
    return (unsigned int)((a & 0xff) | ((b & 0xff) << 8) |
                          ((c & 0xff) << 16) | ((d & 0xff) << 24));
}

__global__ __launch_bounds__(256) void quantize_x_kernel(
    const float* __restrict__ x, unsigned int* __restrict__ xq, int ndwords)
{
    int i = blockIdx.x * blockDim.x + threadIdx.x;
    const int stride = gridDim.x * blockDim.x;
    for (; i < ndwords; i += stride) {
        float4 f = ((const float4*)x)[i];
        xq[i] = qpack4(f, 128.0f, -128, 127);
    }
}

__device__ __forceinline__ v8i load_bfrag_w4(const float* __restrict__ W,
                                             int rowLen, int row, int kcBase) {
    const int lane = threadIdx.x & 31;
    const int g    = lane >> 4;
    const float* base = W + (size_t)row * rowLen + kcBase + g * 8;
    v8i out;
#pragma unroll
    for (int v = 0; v < 4; ++v)
        out[v] = (int)qpack4(*(const float4*)(base + (v >> 1) * 16 + (v & 1) * 4), 8.0f, -8, 7);
#pragma unroll
    for (int v = 0; v < 4; ++v)
        out[4 + v] = (int)qpack4(*(const float4*)(base + 32 + (v >> 1) * 16 + (v & 1) * 4), 8.0f, -8, 7);
    return out;
}

__device__ __forceinline__ v8i load_afrag_i8(const char* __restrict__ base) {
    v8i out;
#pragma unroll
    for (int p = 0; p < 4; ++p) {
        uint2 d = *(const uint2*)(base + p * 16);
        out[2 * p]     = (int)d.x;
        out[2 * p + 1] = (int)d.y;
    }
    return out;
}

__global__ __launch_bounds__(512) void qornn_scan_kernel(
    const char*  __restrict__ xq,
    const float* __restrict__ Wi,
    const float* __restrict__ Wr,
    const float* __restrict__ Wo,
    const float* __restrict__ bvec,
    float* __restrict__ out)
{
    __shared__ char hl[2][B_TILE * H_DIM];

    const int tid   = threadIdx.x;
    const int lane  = tid & 31;
    const int wave  = tid >> 5;
    const int g     = lane >> 4;
    const int n     = lane & 15;
    const int j0    = wave * 16;
    const int bBase = blockIdx.x * B_TILE;

    ((uint2*)hl[0])[tid] = make_uint2(0u, 0u);
    ((uint2*)hl[1])[tid] = make_uint2(0u, 0u);

    v8i bwi = load_bfrag_w4(Wi, I_DIM, j0 + n, 0);
    v8i bwr[4];
#pragma unroll
    for (int kc = 0; kc < 4; ++kc)
        bwr[kc] = load_bfrag_w4(Wr, H_DIM, j0 + n, kc * 64);
    const float bj = bvec[j0 + n];

    const char* xrow = xq + ((size_t)(bBase + n) * T_STEPS) * I_DIM + g * 8;

    v8i ax = load_afrag_i8(xrow);

    __syncthreads();

    for (int t = 0; t < T_STEPS; ++t) {
        const char* hcur = hl[t & 1];
        char*       hnxt = hl[(t & 1) ^ 1];

        v8i acc = {};
        acc = __builtin_amdgcn_wmma_i32_16x16x64_iu8(true, ax, true, bwi, acc,
                                                     false, false);
        const char* hbase = hcur + (n * H_DIM + g * 8);
#pragma unroll
        for (int kc = 0; kc < 4; ++kc) {
            v8i ah = load_afrag_i8(hbase + kc * 64);
            acc = __builtin_amdgcn_wmma_i32_16x16x64_iu8(true, ah, true, bwr[kc],
                                                         acc, false, false);
        }

        const int tn = (t + 1 < T_STEPS) ? (t + 1) : t;
        const char* xnext = xrow + (size_t)tn * I_DIM;
        __builtin_prefetch(xnext + I_DIM, 0, 1);
        ax = load_afrag_i8(xnext);

#pragma unroll
        for (int r = 0; r < 8; ++r) {
            float z = (float)acc[r] * (1.0f / 1024.0f);
            float m = __builtin_fmaxf(__builtin_fabsf(z) + bj, 0.0f);
            float hv = (z > 0.0f) ? m : ((z < 0.0f) ? -m : 0.0f);
            int hq = qclamp(hv, 128.0f, -128, 127);
            hnxt[(r + 8 * g) * H_DIM + j0 + n] = (char)hq;
        }

        __syncthreads();
    }

    if (wave == 0) {
        const char* hfin  = hl[T_STEPS & 1];
        const char* hbase = hfin + (n * H_DIM + g * 8);
        v8i acc = {};
#pragma unroll
        for (int kc = 0; kc < 4; ++kc) {
            v8i ah = load_afrag_i8(hbase + kc * 64);
            v8i bw = load_bfrag_w4(Wo, H_DIM, n, kc * 64);
            acc = __builtin_amdgcn_wmma_i32_16x16x64_iu8(true, ah, true, bw, acc,
                                                         false, false);
        }
#pragma unroll
        for (int r = 0; r < 8; ++r) {
            int bg = bBase + r + 8 * g;
            out[(size_t)bg * O_DIM + n] = (float)acc[r] * (1.0f / 1024.0f);
        }
    }
}

extern "C" void kernel_launch(void* const* d_in, const int* in_sizes, int n_in,
                              void* d_out, int out_size, void* d_ws, size_t ws_size,
                              hipStream_t stream) {
    (void)n_in; (void)out_size; (void)ws_size;
    const float* x  = (const float*)d_in[0];
    const float* Wi = (const float*)d_in[1];
    const float* Wr = (const float*)d_in[2];
    const float* Wo = (const float*)d_in[3];
    const float* b  = (const float*)d_in[4];
    float* out = (float*)d_out;

    const int nx = in_sizes[0];
    const int B  = nx / (T_STEPS * I_DIM);
    char* xq = (char*)d_ws;

    const int ndwords = nx / 4;
    hipLaunchKernelGGL(quantize_x_kernel, dim3(2048), dim3(256), 0, stream,
                       x, (unsigned int*)xq, ndwords);

    dim3 grid((B + B_TILE - 1) / B_TILE);
    dim3 block(512);
    hipLaunchKernelGGL(qornn_scan_kernel, grid, block, 0, stream,
                       xq, Wi, Wr, Wo, b, out);
}
